// HungarianMatcher_12618613916408
// MI455X (gfx1250) — compile-verified, matrix operand order checked by reading, not run on hardware
//
#include <hip/hip_runtime.h>
#include <hip/hip_bf16.h>

typedef __attribute__((ext_vector_type(16))) _Float16     v16h;
typedef __attribute__((ext_vector_type(8)))  float        v8f;
typedef __attribute__((ext_vector_type(4)))  unsigned int u32x4;
typedef __attribute__((ext_vector_type(8)))  int          i32x8;
typedef __attribute__((ext_vector_type(4)))  int          i32x4;

#define BS    8
#define NQ    1024
#define NTGT  128
#define RTOT  (BS * NQ)     // 8192 rows  (bs*nq)
#define CTOT  (BS * NTGT)   // 1024 cols  (bs*ntgt)

// ---------------------------------------------------------------------------
// Kernel 1: cost matrix. One block = 128 rows x 16 cols. 8 waves, each wave
// owns a 16x16 tile. Logits outer-product term runs on the matrix pipe
// (v_wmma_f32_16x16x32_f16, K=1 outer product in K-slot 0); angle and box L1
// terms are fused into the WMMA accumulator layout and stored coalesced.
// The 16-target box tile (256 B, 2D strided) is staged into LDS by the
// Tensor Data Mover (tensor_load_to_lds, TENSORcnt), issued once per block.
// ---------------------------------------------------------------------------
__global__ __launch_bounds__(256)
void hm_cost_matrix_kernel(const float* __restrict__ logits,
                           const float* __restrict__ boxes,
                           const float* __restrict__ angle,
                           const float* __restrict__ tboxes,
                           const float* __restrict__ tangle,
                           const float* __restrict__ tlabels,
                           float* __restrict__ cm)
{
    __shared__ float  s_prob[128];
    __shared__ float  s_ang[128];
    __shared__ float4 s_box[128];
    __shared__ float  s_tl[16];
    __shared__ float  s_ta[16];
    __shared__ float4 s_tb[16];

    const int tid     = threadIdx.x;
    const int rowBase = blockIdx.y * 128;
    const int colBase = blockIdx.x * 16;

#if __has_builtin(__builtin_amdgcn_tensor_load_to_lds) && __has_builtin(__builtin_amdgcn_s_wait_tensorcnt)
    // --- TDM: DMA the 16x4-f32 target-box tile into LDS (one issue / block).
    if (tid == 0) {
        const unsigned long long ga =
            (unsigned long long)(const void*)(tboxes + 4ull * (unsigned)colBase);
        const unsigned lds = (unsigned)(unsigned long long)(void*)&s_tb[0];

        u32x4 g0;
        g0[0] = 1u;                                   // count=1, user D#
        g0[1] = lds;                                  // lds_addr
        g0[2] = (unsigned)(ga & 0xFFFFFFFFull);       // global_addr lo
        g0[3] = (unsigned)((ga >> 32) & 0x1FFFFFFull) // global_addr hi (57-bit)
              | (2u << 30);                           // type = 2 ("image")

        i32x8 g1;
        g1[0] = (int)(2u << 16);                      // wgmask=0, data_size=4B
        g1[1] = (int)(4u << 16);                      // tensor_dim0 = 4 (lo16)
        g1[2] = (int)((0u) | (1024u << 16));          // dim0 hi16=0, tensor_dim1 lo16=1024
        g1[3] = (int)((0u) | (4u << 16));             // dim1 hi16=0, tile_dim0 = 4
        g1[4] = (int)(16u);                           // tile_dim1 = 16, tile_dim2 = 0
        g1[5] = (int)(4u);                            // tensor_dim0_stride = 4 (lo32)
        g1[6] = 0;                                    // stride0 hi16, stride1 lo16
        g1[7] = 0;                                    // stride1 hi32

        i32x4 g2; g2[0] = 0; g2[1] = 0; g2[2] = 0; g2[3] = 0;  // <=2D: unused
        i32x4 g3; g3[0] = 0; g3[1] = 0; g3[2] = 0; g3[3] = 0;
        i32x8 g4;                                     // 6-arg form: extra group
        g4[0] = 0; g4[1] = 0; g4[2] = 0; g4[3] = 0;
        g4[4] = 0; g4[5] = 0; g4[6] = 0; g4[7] = 0;

        __builtin_amdgcn_tensor_load_to_lds(g0, g1, g2, g3, g4, 0);
        __builtin_amdgcn_s_wait_tensorcnt(0);
    }
    const bool tdm_boxes = true;
#else
    const bool tdm_boxes = false;
#endif

    if (tid < 128) {
        const int r = rowBase + tid;
        const float lg = logits[r];
        s_prob[tid] = 1.0f / (1.0f + __expf(-lg));   // sigmoid
        s_ang[tid]  = angle[r];
        s_box[tid]  = ((const float4*)boxes)[r];     // global_load_b128
    } else if (tid >= 128 && tid < 144) {
        const int c = colBase + (tid - 128);
        __builtin_prefetch(&tangle[c], 0, 0);        // global_prefetch_b8
        s_tl[tid - 128] = tlabels[c];
        s_ta[tid - 128] = tangle[c];
        if (!tdm_boxes) s_tb[tid - 128] = ((const float4*)tboxes)[c];
    }
    __syncthreads();

    const int wave  = tid >> 5;          // 8 waves -> 8 row sub-tiles of 16
    const int lane  = tid & 31;
    const int mrow0 = wave * 16;

    // Outer product prob x (-2*label) on the matrix pipe.
    // 16-bit A layout: lanes 0-15 hold M=0..15, h[0] = K=0 -> put prob there.
    // B layout (32x16): lanes 0-15 hold N=0..15, h[0] = K=0 -> put -2*label.
    v16h a, bm;
#pragma unroll
    for (int k = 0; k < 16; ++k) { a[k] = (_Float16)0.0f; bm[k] = (_Float16)0.0f; }
    if (lane < 16) {
        a[0]  = (_Float16)s_prob[mrow0 + lane];
        bm[0] = (_Float16)(-2.0f * s_tl[lane]);
    }
    v8f acc;
#pragma unroll
    for (int k = 0; k < 8; ++k) acc[k] = 0.0f;
    acc = __builtin_amdgcn_wmma_f32_16x16x32_f16(false, a, false, bm,
                                                 (short)0, acc, false, false);

    // C/D layout: lane L -> col N = L%16, rows M = i + 8*(L>=16), i = VGPR idx.
    const int    n   = lane & 15;
    const int    mhi = (lane >> 4) * 8;
    const float  ta  = s_ta[n];
    const float4 tb  = s_tb[n];

#pragma unroll
    for (int i = 0; i < 8; ++i) {
        const int m = mrow0 + mhi + i;
        const float4 rb = s_box[m];
        const float l1 = fabsf(rb.x - tb.x) + fabsf(rb.y - tb.y) +
                         fabsf(rb.z - tb.z) + fabsf(rb.w - tb.w);
        const float val = acc[i] + fabsf(s_ang[m] - ta) + 5.0f * l1;
        cm[(size_t)(rowBase + m) * CTOT + (colBase + n)] = val;
    }
}

// ---------------------------------------------------------------------------
// Kernel 2: Jonker-Volgenant shortest-augmenting-path assignment.
// One block per batch. Transposed problem: n=128 rows (targets),
// m=1024 cols (queries). Column scans parallelized across 256 threads,
// argmin via LDS tree reduction; duals/pointers in LDS. cm stays in L2.
// ---------------------------------------------------------------------------
__global__ __launch_bounds__(256)
void hm_jv_assign_kernel(const float* __restrict__ cm,
                         long long* __restrict__ rowOut,
                         long long* __restrict__ colOut)
{
    const int b   = blockIdx.x;
    const int tid = threadIdx.x;
    const int NTH = 256;
    const int n   = NTGT;   // 128
    const int m   = NQ;     // 1024

    __shared__ float u[NTGT + 1];
    __shared__ float v[NQ + 1];
    __shared__ float minv[NQ + 1];
    __shared__ int   p[NQ + 1];
    __shared__ int   way[NQ + 1];
    __shared__ unsigned char used[NQ + 1];
    __shared__ float redV[256];
    __shared__ int   redJ[256];
    __shared__ int   s_j0;
    __shared__ int   s_done;

    // C(i,j), i=1..n (target t=i-1), j=1..m (query q=j-1):
    //   Cb[(j-1)*CTOT + (i-1)]
    const float* Cb = cm + (size_t)b * NQ * CTOT + (size_t)b * NTGT;

    for (int j = tid; j <= m; j += NTH) { v[j] = 0.0f; p[j] = 0; }
    for (int i = tid; i <= n; i += NTH) u[i] = 0.0f;
    __syncthreads();

    for (int i = 1; i <= n; ++i) {
        for (int j = tid; j <= m; j += NTH) { minv[j] = 3.0e38f; used[j] = 0; }
        if (tid == 0) { p[0] = i; s_j0 = 0; s_done = 0; }
        __syncthreads();

        while (true) {
            const int j0 = s_j0;
            if (tid == 0) used[j0] = 1;
            __syncthreads();
            const int   i0  = p[j0];
            const float ui0 = u[i0];

            // scan free columns: relax minv/way, find argmin
            float lmin = 3.0e38f; int lj = 1;
            for (int j = tid + 1; j <= m; j += NTH) {
                if (!used[j]) {
                    const float cur =
                        Cb[(size_t)(j - 1) * CTOT + (i0 - 1)] - ui0 - v[j];
                    float mv = minv[j];
                    if (cur < mv) { mv = cur; minv[j] = cur; way[j] = j0; }
                    if (mv < lmin) { lmin = mv; lj = j; }
                }
            }
            redV[tid] = lmin; redJ[tid] = lj;
            __syncthreads();
            for (int s = NTH / 2; s > 0; s >>= 1) {
                if (tid < s && redV[tid + s] < redV[tid]) {
                    redV[tid] = redV[tid + s]; redJ[tid] = redJ[tid + s];
                }
                __syncthreads();
            }
            const int   j1    = redJ[0];
            const float delta = redV[0];
            __syncthreads();

            // dual updates (p[j] distinct over used j -> race-free)
            for (int j = tid; j <= m; j += NTH) {
                if (used[j]) { v[j] -= delta; u[p[j]] += delta; }
                else if (j >= 1) minv[j] -= delta;
            }
            __syncthreads();
            if (tid == 0) {
                s_j0   = j1;
                s_done = (p[j1] == 0) ? 1 : 0;
            }
            __syncthreads();
            if (s_done) break;
        }

        if (tid == 0) {  // augment along 'way'
            int j0 = s_j0;
            while (j0 != 0) { const int j1 = way[j0]; p[j0] = p[j1]; j0 = j1; }
        }
        __syncthreads();
    }

    // Emit (query,target) pairs sorted by query index ascending.
    if (tid == 0) {
        int k = 0;
        for (int q = 0; q < m; ++q) {
            const int t = p[q + 1];
            if (t > 0 && k < NTGT) {
                rowOut[b * NTGT + k] = (long long)q;
                colOut[b * NTGT + k] = (long long)(t - 1);
                ++k;
            }
        }
    }
}

// ---------------------------------------------------------------------------
extern "C" void kernel_launch(void* const* d_in, const int* in_sizes, int n_in,
                              void* d_out, int out_size, void* d_ws, size_t ws_size,
                              hipStream_t stream) {
    const float* logits  = (const float*)d_in[0];  // (8,1024,1)
    const float* boxes   = (const float*)d_in[1];  // (8,1024,4)
    const float* angle   = (const float*)d_in[2];  // (8,1024,1)
    const float* tboxes  = (const float*)d_in[3];  // (8,128,4)
    const float* tangle  = (const float*)d_in[4];  // (8,128)
    const float* tlabels = (const float*)d_in[5];  // (8,128)

    float* cm = (float*)d_out;
    const size_t cmBytes = (size_t)RTOT * CTOT * sizeof(float);
    long long* rowOut = (long long*)((char*)d_out + cmBytes);
    long long* colOut = rowOut + (size_t)BS * NTGT;

    dim3 grid(CTOT / 16, RTOT / 128);   // 64 x 64 blocks
    hm_cost_matrix_kernel<<<grid, 256, 0, stream>>>(
        logits, boxes, angle, tboxes, tangle, tlabels, cm);

    hm_jv_assign_kernel<<<BS, 256, 0, stream>>>(cm, rowOut, colOut);
}
